// Model_22247930593573
// MI455X (gfx1250) — matrix operand order repaired and checked by reading, not compiled, not run on hardware
//
#include <hip/hip_runtime.h>
#include <math.h>

// ---------------------------------------------------------------------------
// MI455X / gfx1250 implementation.
// All large linears -> v_wmma_f32_16x16x32_bf16. GEMM v3.1:
//   * each wave computes two 16x64 C strips (8 WMMAs / K-step, A fragments
//     reused 4x, B fragments reused 2x)
//   * packed-B K-step tile staged in double-buffered LDS via
//     GLOBAL_LOAD_ASYNC_TO_LDS_B128 (ASYNCcnt); one barrier per K-step
//   * global_prefetch_b8 for the next A chunks
//   * tile-activity predicates routed through readfirstlane so the compiler
//     emits scalar branches (EXEC stays all-ones at every WMMA, per ISA 7.12)
// Small ops (LN, 21x21 attention, 48-pt DFT) stay on VALU: <1% of FLOPs.
//
// Input leaf order assumes jax-style flatten of the setup_inputs() dict with
// dict keys sorted alphabetically (params before x_enc; lists in order).
// ---------------------------------------------------------------------------

#if defined(__has_builtin)
#if __has_builtin(__builtin_amdgcn_global_load_async_to_lds_b128) && \
    __has_builtin(__builtin_amdgcn_s_wait_asynccnt)
#define USE_ASYNC_LDS 1
#endif
#endif
#ifndef USE_ASYNC_LDS
#define USE_ASYNC_LDS 0
#endif

#if USE_ASYNC_LDS
#define ASG __attribute__((address_space(1)))
#define ASL __attribute__((address_space(3)))
typedef int v4i __attribute__((vector_size(16)));
#endif

typedef __bf16 bf16_t;
typedef __attribute__((ext_vector_type(16))) __bf16 v16bf;
typedef __attribute__((ext_vector_type(8)))  __bf16 v8bf;
typedef __attribute__((ext_vector_type(8)))  float  v8f;

// model constants
constexpr int Bn    = 32;
constexpr int SEQ   = 512;
constexpr int NVv   = 21;
constexpr int D     = 512;
constexpr int Hh    = 8;
constexpr int DKk   = 64;
constexpr int DFF   = 2048;
constexpr int PLEN  = 48;
constexpr int STR   = 24;
constexpr int PRED  = 96;
constexpr int NF    = 25;
constexpr int Pp    = 21;
constexpr int BNt   = Bn * NVv;        // 672
constexpr int Mt    = BNt * Pp;        // 14112 tokens (multiple of 16)
constexpr int HEADK = 2 * D * Pp;      // 21504 (multiple of 32)

__device__ __forceinline__ float dev_sigmoid(float x) { return 1.0f / (1.0f + expf(-x)); }
__device__ __forceinline__ float dev_gelu(float x)    { return 0.5f * x * (1.0f + erff(x * 0.70710678118654752f)); }

// ---------------------------------------------------------------------------
// Weight packing: W is [N,K] row-major fp32 (torch linear weight, y = x W^T).
// B operand tile is 32(K) x 16(N). k order used here (contiguous, same as A loads): lanes 0-15
// hold K=0..15 of column n=lane, lanes 16-31 hold K=16..31 of column n=lane-16.
// Packed layout: [(kt*tilesN + tn)*32 + lane]*16 + j  (j = K within half).
// K is zero-padded to Kpad (multiple of 32).
// ---------------------------------------------------------------------------
__global__ void pack_w_kernel(const float* __restrict__ W, bf16_t* __restrict__ out,
                              int N, int K, int Kpad)
{
    int id = blockIdx.x * 256 + threadIdx.x;
    int total = Kpad * N;
    if (id >= total) return;
    int tile   = id >> 9;          // / 512 elems per tile
    int within = id & 511;
    int lane   = within >> 4;
    int j      = within & 15;
    int tilesN = N >> 4;
    int kt = tile / tilesN;
    int tn = tile % tilesN;
    int n  = tn * 16 + (lane & 15);
    int kk = kt * 32 + (lane >> 4) * 16 + j;
    float v = (kk < K) ? W[(size_t)n * K + kk] : 0.0f;
    out[id] = (bf16_t)v;
}

// ---------------------------------------------------------------------------
// WMMA GEMM v3.1: C[M,N] = A[M,K] * W^T + bias.
//   block = 256 threads = 8 waves; block tile = 256(M) x (NTW*16)(N)
//   wave w -> M-tiles {blk*16+w, blk*16+8+w}, NTW C tiles along N each
// ACT: 0 none, 1 gelu, 2 sigmoid.  WF: write f32 C.  WB: write bf16 C.
// ldc/coff allow writes into column slices (concat fusion).
// ---------------------------------------------------------------------------
template<int ACT, bool BIAS, bool WF, bool WB, int NTW>
__global__ void gemm_bf16_wmma(const bf16_t* __restrict__ A, int lda,
                               const bf16_t* __restrict__ Bp,
                               const float* __restrict__ bias,
                               float* __restrict__ Cf, bf16_t* __restrict__ Cb,
                               int ldc, int coff, int M, int N, int K)
{
    __shared__ __align__(16) bf16_t lb[2][NTW * 512];  // double-buffered B tile
    int tilesM = M >> 4, tilesN = N >> 4;
    int wave = threadIdx.x >> 5;
    int lane = threadIdx.x & 31;
    int tnb  = blockIdx.y * NTW;                       // first n-tile of block
    int tm0  = blockIdx.x * 16 + wave;
    int tm1  = tm0 + 8;
    // wave-uniform activity flags, forced into SGPRs so the guards below are
    // scalar branches (EXEC must be all-ones when a WMMA issues, ISA 7.12)
    int act0 = __builtin_amdgcn_readfirstlane(tm0 < tilesM ? 1 : 0);
    int act1 = __builtin_amdgcn_readfirstlane(tm1 < tilesM ? 1 : 0);
    int r = lane & 15, hlf = lane >> 4;

    const bf16_t* arow0 = A + (size_t)((act0 ? tm0 : 0) * 16 + r) * lda;
    const bf16_t* arow1 = A + (size_t)((act1 ? tm1 : 0) * 16 + r) * lda;
    v8f c0[NTW] = {};
    v8f c1[NTW] = {};
    int KT = K >> 5;

    auto stage = [&](int kt, int buf) {
        const bf16_t* src = Bp + ((size_t)kt * tilesN + tnb) * 512;
        bf16_t* dst = lb[buf];
        constexpr int CH = NTW * 64;                   // # of 16B chunks
        int i = threadIdx.x;
        if (CH == 256 || i < CH) {
#if USE_ASYNC_LDS
            __builtin_amdgcn_global_load_async_to_lds_b128(
                (ASG v4i*)(src + i * 8), (ASL v4i*)(dst + i * 8), 0, 0);
#else
            ((uint4*)dst)[i] = ((const uint4*)src)[i];
#endif
        }
    };

    stage(0, 0);
#if USE_ASYNC_LDS
    __builtin_amdgcn_s_wait_asynccnt(0);
#endif
    __syncthreads();

    for (int kt = 0; kt < KT; ++kt) {
        int cb = kt & 1;
        if (kt + 1 < KT) stage(kt + 1, cb ^ 1);        // stream next B tile
        if (act0 || act1) {
            union { v16bf v; v8bf h[2]; } a0, a1;
            if (act0) {
                a0.h[0] = *(const v8bf*)(arow0 + kt * 32 + hlf * 16);
                a0.h[1] = *(const v8bf*)(arow0 + kt * 32 + hlf * 16 + 8);
                if (kt + 1 < KT) __builtin_prefetch(arow0 + (kt + 1) * 32, 0, 1);
            }
            if (act1) {
                a1.h[0] = *(const v8bf*)(arow1 + kt * 32 + hlf * 16);
                a1.h[1] = *(const v8bf*)(arow1 + kt * 32 + hlf * 16 + 8);
                if (kt + 1 < KT) __builtin_prefetch(arow1 + (kt + 1) * 32, 0, 1);
            }
#pragma unroll
            for (int t = 0; t < NTW; ++t) {
                union { v16bf v; v8bf h[2]; } b;
                const bf16_t* bp = lb[cb] + t * 512 + lane * 16;
                b.h[0] = *(const v8bf*)(bp);
                b.h[1] = *(const v8bf*)(bp + 8);
                if (act0)
                    c0[t] = __builtin_amdgcn_wmma_f32_16x16x32_bf16(
                        false, a0.v, false, b.v, (short)0, c0[t], false, false);
                if (act1)
                    c1[t] = __builtin_amdgcn_wmma_f32_16x16x32_bf16(
                        false, a1.v, false, b.v, (short)0, c1[t], false, false);
            }
        }
#if USE_ASYNC_LDS
        __builtin_amdgcn_s_wait_asynccnt(0);
#endif
        __syncthreads();
    }

    // epilogue
    for (int s = 0; s < 2; ++s) {
        int act = s ? act1 : act0;
        if (!act) continue;
        int tm = s ? tm1 : tm0;
        v8f* c = s ? c1 : c0;
        int rbase = tm * 16 + hlf * 8;                 // C layout: half -> M 0-7 / 8-15
#pragma unroll
        for (int t = 0; t < NTW; ++t) {
            int col  = (tnb + t) * 16 + r;
            float bv = 0.0f;
            if (BIAS) bv = bias[col];
            int outc = col + coff;
#pragma unroll
            for (int i = 0; i < 8; ++i) {
                float v = c[t][i] + bv;
                if (ACT == 1) v = dev_gelu(v);
                else if (ACT == 2) v = dev_sigmoid(v);
                size_t idx = (size_t)(rbase + i) * ldc + outc;
                if (WF) Cf[idx] = v;
                if (WB) Cb[idx] = (bf16_t)v;
            }
        }
    }
}

template<int ACT, bool BIAS, bool WF, bool WB, int NTW>
static inline void launch_gemm(const bf16_t* A, int lda, const bf16_t* W,
                               const float* bias, float* Cf, bf16_t* Cb,
                               int ldc, int coff, int m, int n, int k, hipStream_t s)
{
    dim3 grid((m / 16 + 15) / 16, (n / 16) / NTW);
    gemm_bf16_wmma<ACT, BIAS, WF, WB, NTW><<<grid, 256, 0, s>>>(
        A, lda, W, bias, Cf, Cb, ldc, coff, m, n, k);
}

// ---------------------------------------------------------------------------
// Instance-norm stats over SEQ per (b,nv). x_enc layout [B,SEQ,NV].
// ---------------------------------------------------------------------------
__global__ void stats_kernel(const float* __restrict__ x, float* __restrict__ meanA,
                             float* __restrict__ stdA)
{
    int bn = blockIdx.x;                 // 0..671
    int b = bn / NVv, v = bn % NVv;
    int tid = threadIdx.x;               // 256
    __shared__ float s0[256], s1[256];
    float a0 = x[((size_t)b * SEQ + tid) * NVv + v];
    float a1 = x[((size_t)b * SEQ + tid + 256) * NVv + v];
    s0[tid] = a0 + a1;
    s1[tid] = a0 * a0 + a1 * a1;
    __syncthreads();
    for (int off = 128; off > 0; off >>= 1) {
        if (tid < off) { s0[tid] += s0[tid + off]; s1[tid] += s1[tid + off]; }
        __syncthreads();
    }
    if (tid == 0) {
        float m = s0[0] / SEQ;
        float var = s1[0] / SEQ - m * m;
        meanA[bn] = m;
        stdA[bn]  = sqrtf(var + 1e-5f);
    }
}

// Unfold with replication pad + normalize; emits f32 patches and K=64-padded bf16 GEMM A.
__global__ void patch_kernel(const float* __restrict__ x, const float* __restrict__ meanA,
                             const float* __restrict__ stdA,
                             float* __restrict__ patches, bf16_t* __restrict__ pA)
{
    int id = blockIdx.x * 256 + threadIdx.x;
    if (id >= Mt * 64) return;
    int j   = id & 63;
    int row = id >> 6;                   // bn*P + p
    int p   = row % Pp;
    int bn  = row / Pp;
    if (j < PLEN) {
        int t  = p * STR + j;
        int ts = t < SEQ ? t : SEQ - 1;  // ReplicationPad1d
        int b = bn / NVv, v = bn % NVv;
        float val = (x[((size_t)b * SEQ + ts) * NVv + v] - meanA[bn]) / stdA[bn];
        patches[(size_t)row * PLEN + j] = val;
        pA[id] = (bf16_t)val;
    } else {
        pA[id] = (bf16_t)0.0f;           // K padding 48 -> 64
    }
}

__global__ void posemb_kernel(float* __restrict__ pe)
{
    int id = blockIdx.x * 256 + threadIdx.x;
    if (id >= Pp * D) return;
    int p = id / D, d = id % D;
    int i2 = d & ~1;
    float div = expf((float)i2 * (-9.210340371976184f / (float)D)); // -ln(1e4)/D
    float a = (float)p * div;
    pe[id] = (d & 1) ? cosf(a) : sinf(a);
}

__global__ void addpe_kernel(const float* __restrict__ gin, const float* __restrict__ pe,
                             float* __restrict__ xf, bf16_t* __restrict__ xb)
{
    int id = blockIdx.x * 256 + threadIdx.x;
    if (id >= Mt * D) return;
    int d = id % D;
    int m = id / D;
    float v = gin[id] + pe[(m % Pp) * D + d];
    xf[id] = v;
    xb[id] = (bf16_t)v;
}

// 48-point real DFT (naive): rfft -> re, im, amp  [M, 25]
__global__ void dft_kernel(const float* __restrict__ patches, float* __restrict__ re,
                           float* __restrict__ im, float* __restrict__ amp)
{
    int id = blockIdx.x * 256 + threadIdx.x;
    if (id >= Mt * NF) return;
    int f = id % NF;
    int m = id / NF;
    const float* pr = patches + (size_t)m * PLEN;
    float sr = 0.0f, si = 0.0f;
    float w = 6.283185307179586f * (float)f / (float)PLEN;
#pragma unroll 4
    for (int j = 0; j < PLEN; ++j) {
        float s, c;
        sincosf(w * (float)j, &s, &c);
        float x = pr[j];
        sr += x * c;
        si -= x * s;
    }
    re[id] = sr;
    im[id] = si;
    amp[id] = sqrtf(sr * sr + si * si);
}

// imp = sigmoid(amp @ Wimp^T + b)   (25x25, plain VALU)
__global__ void imp_kernel(const float* __restrict__ amp, const float* __restrict__ W,
                           const float* __restrict__ bias, float* __restrict__ imp)
{
    int id = blockIdx.x * 256 + threadIdx.x;
    if (id >= Mt * NF) return;
    int fo = id % NF;
    int m  = id / NF;
    float s = bias[fo];
#pragma unroll
    for (int fi = 0; fi < NF; ++fi) s += amp[(size_t)m * NF + fi] * W[fo * NF + fi];
    imp[id] = dev_sigmoid(s);
}

// rr = re*imp*filt, ii = im*imp*filt  -> bf16 padded to K=32
__global__ void mod_kernel(const float* __restrict__ re, const float* __restrict__ im,
                           const float* __restrict__ imp, const float* __restrict__ filt,
                           bf16_t* __restrict__ rr, bf16_t* __restrict__ ii)
{
    int id = blockIdx.x * 256 + threadIdx.x;
    if (id >= Mt * 32) return;
    int f = id & 31;
    int m = id >> 5;
    if (f < NF) {
        float g = imp[(size_t)m * NF + f] * filt[f];
        rr[id] = (bf16_t)(re[(size_t)m * NF + f] * g);
        ii[id] = (bf16_t)(im[(size_t)m * NF + f] * g);
    } else {
        rr[id] = (bf16_t)0.0f;
        ii[id] = (bf16_t)0.0f;
    }
}

// LayerNorm over D=512 of (a + gate*b), optional gate/b; writes f32 and/or bf16.
__global__ void ln_kernel(const float* __restrict__ a, const float* __restrict__ b,
                          const float* __restrict__ gate,
                          const float* __restrict__ g, const float* __restrict__ be,
                          float* __restrict__ of, bf16_t* __restrict__ ob, int obld)
{
    int row = blockIdx.x;
    int tid = threadIdx.x;               // 256, 2 elems each
    __shared__ float s0[256], s1[256];
    size_t base = (size_t)row * D;
    float x0 = a[base + tid];
    float x1 = a[base + tid + 256];
    if (b) {
        float b0 = b[base + tid], b1 = b[base + tid + 256];
        if (gate) { b0 *= gate[base + tid]; b1 *= gate[base + tid + 256]; }
        x0 += b0; x1 += b1;
    }
    s0[tid] = x0 + x1;
    s1[tid] = x0 * x0 + x1 * x1;
    __syncthreads();
    for (int off = 128; off > 0; off >>= 1) {
        if (tid < off) { s0[tid] += s0[tid + off]; s1[tid] += s1[tid + off]; }
        __syncthreads();
    }
    float m = s0[0] / D;
    float var = s1[0] / D - m * m;
    float rstd = rsqrtf(var + 1e-5f);
    float y0 = (x0 - m) * rstd * g[tid] + be[tid];
    float y1 = (x1 - m) * rstd * g[tid + 256] + be[tid + 256];
    if (of) { of[base + tid] = y0; of[base + tid + 256] = y1; }
    if (ob) {
        size_t ob_base = (size_t)row * obld;
        ob[ob_base + tid] = (bf16_t)y0;
        ob[ob_base + tid + 256] = (bf16_t)y1;
    }
}

__global__ void ew_add_kernel(float* __restrict__ a, const float* __restrict__ b, int n)
{
    int id = blockIdx.x * 256 + threadIdx.x;
    if (id < n) a[id] += b[id];
}

__global__ void ew_mulsig_kernel(float* __restrict__ a, const float* __restrict__ b, int n)
{
    int id = blockIdx.x * 256 + threadIdx.x;
    if (id < n) a[id] *= dev_sigmoid(b[id]);
}

__global__ void f2b_kernel(const float* __restrict__ a, bf16_t* __restrict__ b, int n)
{
    int id = blockIdx.x * 256 + threadIdx.x;
    if (id < n) b[id] = (bf16_t)a[id];
}

// Tiny attention: L=21, dk=64; one thread per (bn, h, l).
__global__ void attn_kernel(const float* __restrict__ Q, const float* __restrict__ K,
                            const float* __restrict__ V, float* __restrict__ O)
{
    int gid = blockIdx.x * 256 + threadIdx.x;
    if (gid >= BNt * Hh * Pp) return;
    int l  = gid % Pp;
    int h  = (gid / Pp) % Hh;
    int bn = gid / (Pp * Hh);
    size_t rowl = ((size_t)bn * Pp + l) * D + h * DKk;
    float sc[Pp];
    float mx = -1e30f;
    for (int s = 0; s < Pp; ++s) {
        size_t rows = ((size_t)bn * Pp + s) * D + h * DKk;
        float acc = 0.0f;
#pragma unroll 8
        for (int d = 0; d < DKk; ++d) acc += Q[rowl + d] * K[rows + d];
        acc *= 0.125f;                   // 1/sqrt(64)
        sc[s] = acc;
        mx = fmaxf(mx, acc);
    }
    float sum = 0.0f;
    for (int s = 0; s < Pp; ++s) { sc[s] = expf(sc[s] - mx); sum += sc[s]; }
    float inv = 1.0f / sum;
    float acc[DKk];
#pragma unroll
    for (int d = 0; d < DKk; ++d) acc[d] = 0.0f;
    for (int s = 0; s < Pp; ++s) {
        size_t rows = ((size_t)bn * Pp + s) * D + h * DKk;
        float w = sc[s] * inv;
#pragma unroll 8
        for (int d = 0; d < DKk; ++d) acc[d] += w * V[rows + d];
    }
#pragma unroll 8
    for (int d = 0; d < DKk; ++d) O[rowl + d] = acc[d];
}

// FlattenHead gather: headA[bn, c*P + p] = (c<512 ? tf : gf)[bn*P+p, c(%512)]
__global__ void gather_kernel(const float* __restrict__ tf, const float* __restrict__ gf,
                              bf16_t* __restrict__ headA)
{
    int id = blockIdx.x * 256 + threadIdx.x;
    if (id >= BNt * HEADK) return;
    int j  = id % HEADK;
    int bn = id / HEADK;
    int c = j / Pp, p = j % Pp;
    size_t row = (size_t)bn * Pp + p;
    float v = (c < D) ? tf[row * D + c] : gf[row * D + (c - D)];
    headA[id] = (bf16_t)v;
}

// dec[bn, pr] -> out[b, pr, nv] * std + mean
__global__ void final_kernel(const float* __restrict__ dec, const float* __restrict__ stdA,
                             const float* __restrict__ meanA, float* __restrict__ out)
{
    int id = blockIdx.x * 256 + threadIdx.x;
    if (id >= BNt * PRED) return;
    int pr = id % PRED;
    int bn = id / PRED;
    int b = bn / NVv, nv = bn % NVv;
    out[((size_t)b * PRED + pr) * NVv + nv] = dec[id] * stdA[bn] + meanA[bn];
}

// ---------------------------------------------------------------------------
// Host orchestration
// ---------------------------------------------------------------------------
extern "C" void kernel_launch(void* const* d_in, const int* in_sizes, int n_in,
                              void* d_out, int out_size, void* d_ws, size_t ws_size,
                              hipStream_t stream)
{
    (void)in_sizes; (void)n_in; (void)out_size; (void)ws_size;
    auto F = [&](int i) -> const float* { return (const float*)d_in[i]; };

    // ---- workspace bump allocator -----------------------------------------
    size_t cur = 0;
    auto alloc  = [&](size_t bytes) -> void* {
        cur = (cur + 255) & ~(size_t)255;
        void* p = (char*)d_ws + cur;
        cur += bytes;
        return p;
    };
    auto allocF = [&](size_t n) { return (float*)alloc(n * 4); };
    auto allocB = [&](size_t n) { return (bf16_t*)alloc(n * 2); };

    auto packW = [&](const float* W, int N, int K, int Kpad) -> const bf16_t* {
        bf16_t* dst = allocB((size_t)Kpad * N);
        int elems = Kpad * N;
        pack_w_kernel<<<(elems + 255) / 256, 256, 0, stream>>>(W, dst, N, K, Kpad);
        return dst;
    };

    // ---- input leaf indices (sorted-key flatten) --------------------------
    // freq: 0 enc.b, 1 enc.w, 2 enc_ln.b, 3 enc_ln.g, 4 filt, 5 imp.b, 6 imp.w,
    //       7 ip.b, 8 ip.w, 9 rp.b, 10 rp.w
    // guided_layers: base 11 (24/layer); head: 59 b, 60 w
    // inter_layers: base 61 (8/layer); norm_g: 77 b, 78 g; norm_t: 79 b, 80 g
    // time_layers: base 81 (24/layer); val_emb.w: 129; x_enc: 130
    struct LayerP {
        const bf16_t *q,*k,*v,*qb,*kg,*vb,*o,*c1,*c2;
        const float *qB,*kB,*vB,*qbB,*kgB,*vbB,*oB,*c1B,*c2B;
        const float *normG,*normB,*n1G,*n1B,*n2G,*n2B;
    };
    LayerP L[4];
    const int bases[4] = { 81, 81 + 24, 11, 11 + 24 };   // time0,time1,guided0,guided1
    for (int i = 0; i < 4; ++i) {
        int b = bases[i];
        L[i].kB  = F(b + 0);  L[i].k  = packW(F(b + 1),  D,   D,   D);
        L[i].kgB = F(b + 2);  L[i].kg = packW(F(b + 3),  D,   D,   D);
        L[i].normB = F(b + 4); L[i].normG = F(b + 5);
        L[i].oB  = F(b + 6);  L[i].o  = packW(F(b + 7),  D,   D,   D);
        L[i].qB  = F(b + 8);  L[i].q  = packW(F(b + 9),  D,   D,   D);
        L[i].qbB = F(b + 10); L[i].qb = packW(F(b + 11), D,   D,   D);
        L[i].vB  = F(b + 12); L[i].v  = packW(F(b + 13), D,   D,   D);
        L[i].vbB = F(b + 14); L[i].vb = packW(F(b + 15), D,   D,   D);
        L[i].c1B = F(b + 16); L[i].c1 = packW(F(b + 17), DFF, D,   D);
        L[i].c2B = F(b + 18); L[i].c2 = packW(F(b + 19), D,   DFF, DFF);
        L[i].n1B = F(b + 20); L[i].n1G = F(b + 21);
        L[i].n2B = F(b + 22); L[i].n2G = F(b + 23);
    }
    struct InterP { const bf16_t *gt,*gg; const float *gtB,*ggB,*ntG,*ntB,*ngG,*ngB; };
    InterP IP[2];
    for (int i = 0; i < 2; ++i) {
        int b = 61 + 8 * i;
        IP[i].ggB = F(b + 0); IP[i].gg = packW(F(b + 1), D, 2 * D, 2 * D);
        IP[i].gtB = F(b + 2); IP[i].gt = packW(F(b + 3), D, 2 * D, 2 * D);
        IP[i].ngB = F(b + 4); IP[i].ngG = F(b + 5);
        IP[i].ntB = F(b + 6); IP[i].ntG = F(b + 7);
    }
    const bf16_t* Wenc  = packW(F(1),   D,    D,     D);
    const bf16_t* Wip   = packW(F(8),   256,  NF,    32);
    const bf16_t* Wrp   = packW(F(10),  256,  NF,    32);
    const bf16_t* Wval  = packW(F(129), D,    PLEN,  64);
    const bf16_t* Whead = packW(F(60),  PRED, HEADK, HEADK);
    const float*  xenc  = F(130);

    // ---- activations ------------------------------------------------------
    float*  meanA  = allocF(BNt);
    float*  stdA   = allocF(BNt);
    float*  patches= allocF((size_t)Mt * PLEN);
    bf16_t* pA     = allocB((size_t)Mt * 64);
    float*  pe     = allocF(Pp * D);
    float*  fre    = allocF((size_t)Mt * NF);
    float*  fim    = allocF((size_t)Mt * NF);
    float*  famp   = allocF((size_t)Mt * NF);
    float*  fimp   = allocF((size_t)Mt * NF);
    bf16_t* rrb    = allocB((size_t)Mt * 32);
    bf16_t* iib    = allocB((size_t)Mt * 32);
    float*  xt     = allocF((size_t)Mt * D);
    float*  xg     = allocF((size_t)Mt * D);
    float*  TN     = allocF((size_t)Mt * D);
    float*  GN     = allocF((size_t)Mt * D);
    bf16_t* xtb    = allocB((size_t)Mt * D);
    bf16_t* xgb    = allocB((size_t)Mt * D);
    bf16_t* x2b    = allocB((size_t)Mt * D);
    bf16_t* abf    = allocB((size_t)Mt * D);
    bf16_t* comb   = allocB((size_t)Mt * 2 * D);
    bf16_t* h1b    = allocB((size_t)Mt * DFF);
    bf16_t* headA  = allocB((size_t)BNt * HEADK);
    float*  T0 = allocF((size_t)Mt * D);
    float*  T1 = allocF((size_t)Mt * D);
    float*  T2 = allocF((size_t)Mt * D);
    float*  T3 = allocF((size_t)Mt * D);
    float*  T4 = allocF((size_t)Mt * D);
    float*  T5 = allocF((size_t)Mt * D);
    float*  decb = allocF((size_t)BNt * PRED);

    // ---- GEMM launch helpers ----------------------------------------------
    auto g_b = [&](const bf16_t* A, int lda, const bf16_t* W, const float* bias,
                   float* C, int ldc, int coff, int m, int n, int k) {
        launch_gemm<0, true, true, false, 4>(A, lda, W, bias, C, nullptr, ldc, coff,
                                             m, n, k, stream);
    };
    auto g_nb = [&](const bf16_t* A, int lda, const bf16_t* W, float* C, int ldc,
                    int m, int n, int k) {
        launch_gemm<0, false, true, false, 4>(A, lda, W, nullptr, C, nullptr, ldc, 0,
                                              m, n, k, stream);
    };
    auto g_gel = [&](const bf16_t* A, int lda, const bf16_t* W, const float* bias,
                     float* C, int ldc, int m, int n, int k) {
        launch_gemm<1, true, true, false, 4>(A, lda, W, bias, C, nullptr, ldc, 0,
                                             m, n, k, stream);
    };
    auto g_gelb = [&](const bf16_t* A, int lda, const bf16_t* W, const float* bias,
                      bf16_t* C, int ldc, int m, int n, int k) {
        launch_gemm<1, true, false, true, 4>(A, lda, W, bias, nullptr, C, ldc, 0,
                                             m, n, k, stream);
    };
    auto g_sig = [&](const bf16_t* A, int lda, const bf16_t* W, const float* bias,
                     float* C, int ldc, int m, int n, int k) {
        launch_gemm<2, true, true, false, 4>(A, lda, W, bias, C, nullptr, ldc, 0,
                                             m, n, k, stream);
    };
    auto g_bb = [&](const bf16_t* A, int lda, const bf16_t* W, const float* bias,
                    bf16_t* C, int ldc, int coff, int m, int n, int k) {
        launch_gemm<0, true, false, true, 4>(A, lda, W, bias, nullptr, C, ldc, coff,
                                             m, n, k, stream);
    };
    auto g_head = [&](const bf16_t* A, int lda, const bf16_t* W, const float* bias,
                      float* C, int ldc, int m, int n, int k) {
        launch_gemm<0, true, true, false, 2>(A, lda, W, bias, C, nullptr, ldc, 0,
                                             m, n, k, stream);      // N=96 -> NTW=2
    };
    auto ln = [&](const float* a, const float* b, const float* gate,
                  const float* g, const float* be, float* of, bf16_t* ob, int obld) {
        ln_kernel<<<Mt, 256, 0, stream>>>(a, b, gate, g, be, of, ob, obld);
    };
    const int EWN = Mt * D;
    const int EWB = (EWN + 255) / 256;

    // ---- front end --------------------------------------------------------
    stats_kernel<<<BNt, 256, 0, stream>>>(xenc, meanA, stdA);
    patch_kernel<<<(Mt * 64 + 255) / 256, 256, 0, stream>>>(xenc, meanA, stdA, patches, pA);
    posemb_kernel<<<(Pp * D + 255) / 256, 256, 0, stream>>>(pe);

    // patch embedding (K=48 padded to 64) + positional
    g_nb(pA, 64, Wval, T0, D, Mt, D, 64);
    addpe_kernel<<<EWB, 256, 0, stream>>>(T0, pe, xt, xtb);

    // freq branch
    dft_kernel<<<(Mt * NF + 255) / 256, 256, 0, stream>>>(patches, fre, fim, famp);
    imp_kernel<<<(Mt * NF + 255) / 256, 256, 0, stream>>>(famp, F(6), F(5), fimp);
    mod_kernel<<<(Mt * 32 + 255) / 256, 256, 0, stream>>>(fre, fim, fimp, F(4), rrb, iib);
    g_bb(rrb, 32, Wrp, F(9), abf, D, 0,   Mt, 256, 32);   // feat[:, :256]
    g_bb(iib, 32, Wip, F(7), abf, D, 256, Mt, 256, 32);   // feat[:, 256:]
    g_gel(abf, D, Wenc, F(0), T2, D, Mt, D, D);
    ln(T2, nullptr, nullptr, F(3), F(2), xg, xgb, D);     // enc_ln -> guided feat

    // ---- dual encoder layers ----------------------------------------------
    auto enc_layer = [&](const float* x, const bf16_t* xb, const bf16_t* ctxb,
                         const LayerP& lp, float* outF, bf16_t* outB, int outBld) {
        // modulated attention
        g_b(xb,   D, lp.q,  lp.qB,  T0, D, 0, Mt, D, D);
        g_b(ctxb, D, lp.qb, lp.qbB, T1, D, 0, Mt, D, D);
        ew_add_kernel<<<EWB, 256, 0, stream>>>(T0, T1, EWN);          // Qm
        g_b(xb,   D, lp.k,  lp.kB,  T1, D, 0, Mt, D, D);
        g_b(ctxb, D, lp.kg, lp.kgB, T2, D, 0, Mt, D, D);
        ew_mulsig_kernel<<<EWB, 256, 0, stream>>>(T1, T2, EWN);       // Km
        g_b(xb,   D, lp.v,  lp.vB,  T2, D, 0, Mt, D, D);
        g_b(ctxb, D, lp.vb, lp.vbB, T3, D, 0, Mt, D, D);
        ew_add_kernel<<<EWB, 256, 0, stream>>>(T2, T3, EWN);          // Vm
        attn_kernel<<<(BNt * Hh * Pp + 255) / 256, 256, 0, stream>>>(T0, T1, T2, T3);
        f2b_kernel<<<EWB, 256, 0, stream>>>(T3, abf, EWN);
        g_b(abf, D, lp.o, lp.oB, T4, D, 0, Mt, D, D);
        ln(x, T4, nullptr, lp.normG, lp.normB, T5, nullptr, 0);       // new_x
        // FFN
        ln(x, T5, nullptr, lp.n1G, lp.n1B, T0, x2b, D);               // x2
        g_gelb(x2b, D, lp.c1, lp.c1B, h1b, DFF, Mt, DFF, D);
        g_b(h1b, DFF, lp.c2, lp.c2B, T1, D, 0, Mt, D, DFF);
        ln(T0, T1, nullptr, lp.n2G, lp.n2B, outF, outB, outBld);
    };

    for (int e = 0; e < 2; ++e) {
        enc_layer(xt, xtb, xgb, L[e],     TN, comb,     2 * D);       // t_new -> comb[:, :512]
        enc_layer(xg, xgb, xtb, L[2 + e], GN, comb + D, 2 * D);       // g_new -> comb[:, 512:]
        g_sig(comb, 2 * D, IP[e].gt, IP[e].gtB, T0, D, Mt, D, 2 * D); // gt
        g_sig(comb, 2 * D, IP[e].gg, IP[e].ggB, T1, D, Mt, D, 2 * D); // gg
        ln(TN, GN, T0, IP[e].ntG, IP[e].ntB, xt, xtb, D);             // time_feat
        ln(GN, TN, T1, IP[e].ngG, IP[e].ngB, xg, xgb, D);             // guided_feat
    }

    // ---- head -------------------------------------------------------------
    ln(xt, nullptr, nullptr, F(80), F(79), T0, nullptr, 0);           // norm_t
    ln(xg, nullptr, nullptr, F(78), F(77), T1, nullptr, 0);           // norm_g
    gather_kernel<<<(BNt * HEADK + 255) / 256, 256, 0, stream>>>(T0, T1, headA);
    g_head(headA, HEADK, Whead, F(59), decb, PRED, BNt, PRED, HEADK);
    final_kernel<<<(BNt * PRED + 255) / 256, 256, 0, stream>>>(decb, stdA, meanA,
                                                               (float*)d_out);
}
